// StyleGAN2GeneratorClean_16003048145108
// MI455X (gfx1250) — matrix operand order repaired and checked by reading, not compiled, not run on hardware
//
#include <hip/hip_runtime.h>
#include <hip/hip_bf16.h>
#include <cstdint>
#include <cstddef>

// ---------------------------------------------------------------------------
// Types for CDNA5 WMMA fragments
// ---------------------------------------------------------------------------
typedef unsigned short u16;
typedef __attribute__((ext_vector_type(16))) __bf16        v16bf;
typedef __attribute__((ext_vector_type(8)))  float         v8f;
typedef __attribute__((ext_vector_type(2)))  float         v2f;
typedef __attribute__((ext_vector_type(4)))  float         v4f;
typedef __attribute__((ext_vector_type(4)))  unsigned int  v4u;

union Frag16 { v16bf bf; v4u u[2]; };

__device__ __forceinline__ u16 f2bf(float f) {
  unsigned u = __float_as_uint(f);
  u += 0x7FFFu + ((u >> 16) & 1u);          // round-to-nearest-even
  return (u16)(u >> 16);
}

// bilinear 2x upsample, align_corners=False (half-pixel centers), edge clamp
__device__ __forceinline__ float bilin_up2(const float* __restrict__ img,
                                           int H, int W, int oy, int ox) {
  float cy = (oy + 0.5f) * 0.5f - 0.5f;
  float cx = (ox + 0.5f) * 0.5f - 0.5f;
  float fy = floorf(cy), fx = floorf(cx);
  int y0 = (int)fy, x0 = (int)fx;
  float ty = cy - fy, tx = cx - fx;
  int y1 = min(y0 + 1, H - 1), x1 = min(x0 + 1, W - 1);
  y0 = max(y0, 0); x0 = max(x0, 0);
  float a = img[y0 * W + x0], b = img[y0 * W + x1];
  float c = img[y1 * W + x0], d = img[y1 * W + x1];
  float top = a + tx * (b - a);
  float bot = c + tx * (d - c);
  return top + ty * (bot - top);
}

// ---------------------------------------------------------------------------
// 1) NormStyleCode: x = styles * rsqrt(mean(styles^2) + 1e-8)
// ---------------------------------------------------------------------------
__global__ void __launch_bounds__(512) norm_kernel(const float* __restrict__ styles,
                                                   float* __restrict__ out) {
  __shared__ float red[512];
  int b = blockIdx.x, t = threadIdx.x;
  float v = styles[b * 512 + t];
  red[t] = v * v;
  __syncthreads();
  for (int s = 256; s > 0; s >>= 1) {
    if (t < s) red[t] += red[t + s];
    __syncthreads();
  }
  float mean = red[0] * (1.0f / 512.0f);
  out[b * 512 + t] = v * rsqrtf(mean + 1e-8f);
}

// ---------------------------------------------------------------------------
// 2) MLP layer: [16(pad),512] @ W^T [512,512] + b, leaky(0.2).
//    fp32 WMMA 16x16x4, one wave per 16-column tile.
//    Latent buffers are padded to 16 rows; rows 8..15 are garbage but WMMA
//    row m of D depends only on row m of A, so real rows 0..7 stay exact.
//    All loads/stores unconditional -> zero exec-mask divergence.
// ---------------------------------------------------------------------------
__global__ void __launch_bounds__(32) mlp_wmma_kernel(const float* __restrict__ xin,  // [16,512]
                                                      const float* __restrict__ Wt,   // [512,512]
                                                      const float* __restrict__ bias, // [512]
                                                      float* __restrict__ xout) {     // [16,512]
  const int lane = threadIdx.x & 31;
  const int n0 = blockIdx.x * 16;
  const int nn = lane & 15;
  const int hi = (lane & 16) ? 2 : 0;      // A/B: lanes>=16 hold K+2,K+3
  const float* arow = xin + nn * 512;      // A row (M = lane&15)
  const float* brow = Wt + (size_t)(n0 + nn) * 512;  // B column (N = lane&15)
  v8f acc = {0.f, 0.f, 0.f, 0.f, 0.f, 0.f, 0.f, 0.f};
  for (int k0 = 0; k0 < 512; k0 += 4) {
    v2f a = *(const v2f*)(arow + k0 + hi);
    v2f b = *(const v2f*)(brow + k0 + hi);
    acc = __builtin_amdgcn_wmma_f32_16x16x4_f32(false, a, false, b,
                                                (short)0, acc, false, false);
  }
  // C/D layout: VGPR r -> M = r (+8 for lanes>=16), N = lane&15. Write all 16 rows.
  const int mo = (lane & 16) ? 8 : 0;
  const float bv = bias[n0 + nn];
#pragma unroll
  for (int r = 0; r < 8; ++r) {
    float v = acc[r] + bv;
    v = v > 0.f ? v : 0.2f * v;
    xout[(r + mo) * 512 + n0 + nn] = v;
  }
}

// ---------------------------------------------------------------------------
// 3) Style GEMV: s[b,ci] = latent[b] . mod_W[ci] + mod_b[ci]  (float4 loads)
// ---------------------------------------------------------------------------
__global__ void __launch_bounds__(256) style_kernel(const float* __restrict__ lat,
                                                    const float* __restrict__ modW,
                                                    const float* __restrict__ modB,
                                                    float* __restrict__ s, int CI) {
  int t = blockIdx.x * 256 + threadIdx.x;
  if (t >= 8 * CI) return;
  int b = t / CI, ci = t % CI;
  const v4f* lp = (const v4f*)(lat + b * 512);
  const v4f* wp = (const v4f*)(modW + (size_t)ci * 512);
  float acc = 0.f;
  for (int k = 0; k < 128; ++k) {
    v4f l = lp[k], w = wp[k];
    acc += l.x * w.x + l.y * w.y + l.z * w.z + l.w * w.w;
  }
  s[t] = acc + modB[ci];
}

// ---------------------------------------------------------------------------
// 4) Weight prep: base weight [co,ci,3,3] fp32 -> A [co,tap,ci] bf16,
//    plus wsq[co,ci] = sum_k w^2 (for demod).
// ---------------------------------------------------------------------------
__global__ void __launch_bounds__(256) wprep_kernel(const float* __restrict__ w,
                                                    u16* __restrict__ Ab,
                                                    float* __restrict__ wsq,
                                                    int CO, int CI) {
  int t = blockIdx.x * 256 + threadIdx.x;
  if (t >= CO * CI) return;
  int co = t / CI, ci = t % CI;
  const float* wp = w + (size_t)t * 9;
  float ss = 0.f;
#pragma unroll
  for (int k = 0; k < 9; ++k) {
    float v = wp[k];
    ss += v * v;
    Ab[((size_t)co * 9 + k) * CI + ci] = f2bf(v);
  }
  wsq[t] = ss;
}

// demod[b,co] = rsqrt( sum_ci wsq[co,ci] * s[b,ci]^2 + 1e-8 )
__global__ void __launch_bounds__(256) demod_kernel(const float* __restrict__ s,
                                                    const float* __restrict__ wsq,
                                                    float* __restrict__ dm,
                                                    int CO, int CI) {
  int t = blockIdx.x * 256 + threadIdx.x;
  if (t >= 8 * CO) return;
  int b = t / CO, co = t % CO;
  const float* sp = s + b * CI;
  const float* wp = wsq + (size_t)co * CI;
  float acc = 0.f;
  for (int ci = 0; ci < CI; ++ci) acc += wp[ci] * sp[ci] * sp[ci];
  dm[t] = rsqrtf(acc + 1e-8f);
}

// ---------------------------------------------------------------------------
// 5) Input scaling (modulation folded into activations): xs = bf16(x * s[b,ci])
// ---------------------------------------------------------------------------
__global__ void __launch_bounds__(256) scale_kernel(const float* __restrict__ x,
                                                    int xbstride,  // CI*HW or 0 (const bcast)
                                                    const float* __restrict__ s,
                                                    u16* __restrict__ out,
                                                    int CI, int HW) {
  long t = (long)blockIdx.x * 256 + threadIdx.x;
  long total = (long)8 * CI * HW;
  if (t >= total) return;
  int p = (int)(t % HW);
  long r = t / HW;
  int ci = (int)(r % CI);
  int b = (int)(r / CI);
  float v = x[(size_t)b * xbstride + (size_t)ci * HW + p] * s[b * CI + ci];
  out[t] = f2bf(v);
}

// fused 2x bilinear upsample + modulation scale -> bf16
__global__ void __launch_bounds__(256) up2scale_kernel(const float* __restrict__ x,
                                                       const float* __restrict__ s,
                                                       u16* __restrict__ out,
                                                       int CI, int Hin, int Win) {
  int OH = 2 * Hin, OW = 2 * Win;
  long t = (long)blockIdx.x * 256 + threadIdx.x;
  long total = (long)8 * CI * OH * OW;
  if (t >= total) return;
  int ox = (int)(t % OW);
  long r = t / OW;
  int oy = (int)(r % OH); r /= OH;
  int ci = (int)(r % CI);
  int b = (int)(r / CI);
  const float* img = x + ((size_t)(b * CI + ci)) * Hin * Win;
  float v = bilin_up2(img, Hin, Win, oy, ox) * s[b * CI + ci];
  out[t] = f2bf(v);
}

// ---------------------------------------------------------------------------
// 6) Modulated 3x3 conv as implicit-GEMM with bf16 WMMA 16x16x32.
//    Block = 4 waves: 64 output channels x (NT*16) pixels.
//    B (im2col) tile double-buffered in LDS; 1 barrier per K-step.
//    Each wave shares one A fragment across NT WMMAs (2x math per A load).
//    Epilogue fuses demod*sqrt(2) + noise + bias + leakyReLU(0.2).
// ---------------------------------------------------------------------------
template <int NT>
__global__ void __launch_bounds__(128) conv_wmma_kernel(
    const u16* __restrict__ A,        // [CO,9,CI] bf16
    const u16* __restrict__ xs,       // [8,CI,H,W] bf16 (pre-modulated)
    float* __restrict__ out,          // [8,CO,H,W]
    const float* __restrict__ demod,  // [8,CO]
    const float* __restrict__ noise,  // [H*W]
    const float* __restrict__ noise_w,// [1]
    const float* __restrict__ bias,   // [CO]
    int CO, int CI, int H, int W) {
  const int lane = threadIdx.x & 31;
  const int wave = threadIdx.x >> 5;
  const int HW = H * W;
  const int p0 = blockIdx.x * (16 * NT);
  const int co0 = blockIdx.y * 64 + wave * 16;
  const int b = blockIdx.z;

  __shared__ __align__(16) u16 Bt[2][NT * 16 * 32];  // [buf][n][kci]

  v8f acc[NT];
#pragma unroll
  for (int t = 0; t < NT; ++t) acc[t] = (v8f){0.f,0.f,0.f,0.f,0.f,0.f,0.f,0.f};

  const int m  = co0 + (lane & 15);
  const int ka = (lane & 16) ? 8 : 0;    // A frag half-wave K sub-offset
  const int kb = (lane & 16) ? 16 : 0;   // unused; B frag reads use ka
  const int nn = lane & 15;

  // per-thread im2col constants: column index is invariant across fill loop
  const int lg = (NT == 2) ? 5 : 4;
  const int nfix = threadIdx.x & ((1 << lg) - 1);
  const int kbase = threadIdx.x >> lg;
  const int kstep = 128 >> lg;
  const int pfix = p0 + nfix;
  const int pyy = pfix / W, pxx = pfix % W;

  const int nch = CI >> 5;               // K-chunks per tap
  const int TOT = 9 * nch;

  auto fill = [&](int it, int buf) {
    int tap = it / nch;
    int ci0 = (it - tap * nch) << 5;
    int kh = tap / 3 - 1, kw = tap % 3 - 1;
    int py = pyy + kh, px = pxx + kw;
    bool ok = (py >= 0) & (py < H) & (px >= 0) & (px < W);
    int pyc = min(max(py, 0), H - 1);
    int pxc = min(max(px, 0), W - 1);
    const u16* src = xs + ((size_t)(b * CI + ci0)) * HW + pyc * W + pxc;
    u16* dst = &Bt[buf][0];
#pragma unroll
    for (int i = 0; i < 4 * NT; ++i) {
      int kci = kbase + kstep * i;
      u16 v = src[(size_t)kci * HW];     // unconditional (clamped) load
      if (!ok) v = 0;                    // v_cndmask, no exec branching
      dst[nfix * 32 + kci] = v;
    }
    // prefetch this iteration's A fragment (consumed on the next compute step)
    __builtin_prefetch(A + ((size_t)m * 9 + tap) * CI + ci0 + ka, 0, 0);
  };

  fill(0, 0);
  __syncthreads();

  for (int it = 0; it < TOT; ++it) {
    const int buf = it & 1;
    const int tap = it / nch;
    const int ci0 = (it - tap * nch) << 5;

    // A fragment: 16-bit A 16x32 layout — lane<16: K 0..7 & 16..23; lane>=16: +8
    const u16* ap = A + ((size_t)m * 9 + tap) * CI + ci0;
    Frag16 af;
    af.u[0] = *(const v4u*)(ap + ka);
    af.u[1] = *(const v4u*)(ap + 16 + ka);

#pragma unroll
    for (int t = 0; t < NT; ++t) {
      // B fragment: lanes 0-15 hold K 0..7 & 16..23 for column N; lanes 16-31: +8
      Frag16 bfr;
      const u16* bp = &Bt[buf][(t * 16 + nn) * 32 + ka];
      bfr.u[0] = *(const v4u*)(bp);
      bfr.u[1] = *(const v4u*)(bp + 16);
      acc[t] = __builtin_amdgcn_wmma_f32_16x16x32_bf16(false, af.bf, false, bfr.bf,
                                                       (short)0, acc[t], false, false);
    }

    if (it + 1 < TOT) fill(it + 1, buf ^ 1);
    __syncthreads();
  }

  // Epilogue. C layout: VGPR r -> M = r (+8 for lanes>=16), N = lane&15.
  const int mo = (lane & 16) ? 8 : 0;
  const float nw = noise_w[0];
#pragma unroll
  for (int t = 0; t < NT; ++t) {
    const int p = p0 + t * 16 + nn;
    const float nz = nw * noise[p];
#pragma unroll
    for (int r = 0; r < 8; ++r) {
      int mm = co0 + r + mo;
      float v = acc[t][r] * demod[b * CO + mm] * 1.41421356237f + nz + bias[mm];
      v = v > 0.f ? v : 0.2f * v;
      out[((size_t)(b * CO + mm)) * HW + p] = v;
    }
  }
}

// ---------------------------------------------------------------------------
// 7) to_rgb: 1x1 modulated conv (no demod), co=3, + bias + upsampled skip.
// ---------------------------------------------------------------------------
__global__ void __launch_bounds__(256) rgbw_kernel(const float* __restrict__ w, // [3,CI]
                                                   const float* __restrict__ s, // [8,CI]
                                                   float* __restrict__ wm,      // [8,3,CI]
                                                   int CI) {
  int t = blockIdx.x * 256 + threadIdx.x;
  if (t >= 8 * 3 * CI) return;
  int b = t / (3 * CI);
  int r = t % (3 * CI);
  int c = r / CI, ci = r % CI;
  wm[t] = w[(size_t)c * CI + ci] * s[b * CI + ci];
}

__global__ void __launch_bounds__(256) torgb_kernel(const float* __restrict__ x,
                                                    const float* __restrict__ wm,   // [8,3,CI]
                                                    const float* __restrict__ bias, // [3]
                                                    const float* __restrict__ skipPrev, // [8,3,H/2,W/2] or null
                                                    float* __restrict__ out,        // [8,3,H,W]
                                                    int CI, int H, int W) {
  int HW = H * W;
  int p = blockIdx.x * 256 + threadIdx.x;
  int b = blockIdx.y;
  if (p >= HW) return;
  const float* xp = x + (size_t)b * CI * HW + p;
  const float* w0 = wm + (size_t)(b * 3 + 0) * CI;
  const float* w1 = w0 + CI;
  const float* w2 = w1 + CI;
  float a0 = 0.f, a1 = 0.f, a2 = 0.f;
  for (int ci = 0; ci < CI; ++ci) {
    float xv = xp[(size_t)ci * HW];
    a0 += w0[ci] * xv;
    a1 += w1[ci] * xv;
    a2 += w2[ci] * xv;
  }
  float res[3] = {a0 + bias[0], a1 + bias[1], a2 + bias[2]};
  if (skipPrev) {
    int y = p / W, xc = p % W;
    int Hh = H / 2, Wh = W / 2, HWh = Hh * Wh;
#pragma unroll
    for (int c = 0; c < 3; ++c)
      res[c] += bilin_up2(skipPrev + (size_t)(b * 3 + c) * HWh, Hh, Wh, y, xc);
  }
#pragma unroll
  for (int c = 0; c < 3; ++c)
    out[(size_t)(b * 3 + c) * HW + p] = res[c];
}

// ---------------------------------------------------------------------------
// Host orchestration
// ---------------------------------------------------------------------------
namespace {
struct SC { const float *modW, *modB, *w, *nw, *bias; };   // style conv
struct RG { const float *modW, *modB, *w, *bias; };        // to_rgb

inline unsigned cdiv(long a, long b) { return (unsigned)((a + b - 1) / b); }
}

extern "C" void kernel_launch(void* const* d_in, const int* in_sizes, int n_in,
                              void* d_out, int out_size, void* d_ws, size_t ws_size,
                              hipStream_t stream) {
  (void)in_sizes; (void)n_in; (void)out_size; (void)ws_size;
  const float* const* in = (const float* const*)d_in;

  // ---- input index map (setup_inputs dict order) ----
  const float* styles = in[0];
  const float* mlpW[8]; const float* mlpB[8];
  for (int i = 0; i < 8; ++i) { mlpW[i] = in[1 + 2 * i]; mlpB[i] = in[2 + 2 * i]; }
  const float* constv = in[17];
  SC conv1 = { in[18], in[19], in[20], in[21], in[22] };
  RG rgb1  = { in[23], in[24], in[25], in[26] };
  SC convs[12];
  for (int k = 0; k < 12; ++k) {
    int b0 = 27 + 5 * k;
    convs[k] = { in[b0], in[b0 + 1], in[b0 + 2], in[b0 + 3], in[b0 + 4] };
  }
  RG rgbs[6];
  for (int k = 0; k < 6; ++k) {
    int b0 = 87 + 4 * k;
    rgbs[k] = { in[b0], in[b0 + 1], in[b0 + 2], in[b0 + 3] };
  }
  const float* noises[13];
  for (int l = 0; l < 13; ++l) noises[l] = in[111 + l];

  // ---- workspace layout ----
  char* cur = (char*)d_ws;
  auto alloc = [&](size_t bytes) -> void* {
    void* p = cur;
    cur += (bytes + 255) & ~(size_t)255;
    return p;
  };
  float* lat0  = (float*)alloc(16 * 512 * 4);   // padded to 16 rows for WMMA
  float* lat1  = (float*)alloc(16 * 512 * 4);
  float* sbuf  = (float*)alloc(8 * 512 * 4);
  float* srgb  = (float*)alloc(8 * 512 * 4);
  float* dmbuf = (float*)alloc(8 * 512 * 4);
  u16*   Abuf  = (u16*)alloc((size_t)512 * 9 * 512 * 2);
  float* wsq   = (float*)alloc((size_t)512 * 512 * 4);
  float* wmrgb = (float*)alloc(8 * 3 * 512 * 4);
  u16*   xsbuf = (u16*)alloc((size_t)8 * 256 * 256 * 256 * 2);  // 268 MB (max mod-input)
  float* actA  = (float*)alloc((size_t)8 * 128 * 256 * 256 * 4);// 268 MB (x)
  float* actB  = (float*)alloc((size_t)8 * 128 * 256 * 256 * 4);// 268 MB (tmp)
  float* skipA = (float*)alloc((size_t)8 * 3 * 256 * 256 * 4);
  float* skipB = (float*)alloc((size_t)8 * 3 * 256 * 256 * 4);

  // ---- 1. style MLP ----
  norm_kernel<<<8, 512, 0, stream>>>(styles, lat0);
  float* a = lat0; float* bprev = lat1;
  for (int i = 0; i < 8; ++i) {
    mlp_wmma_kernel<<<32, 32, 0, stream>>>(a, mlpW[i], mlpB[i], bprev);
    float* t = a; a = bprev; bprev = t;
  }
  float* lat = a;  // final latent [8,512] in rows 0..7 (same for all 14 slots)

  auto run_style = [&](const float* mW, const float* mB, float* sdst, int CI) {
    style_kernel<<<cdiv(8L * CI, 256), 256, 0, stream>>>(lat, mW, mB, sdst, CI);
  };
  auto run_conv = [&](const SC& c, const float* noise, const u16* xs,
                      float* y, int CO, int CI, int H, int W) {
    wprep_kernel<<<cdiv((long)CO * CI, 256), 256, 0, stream>>>(c.w, Abuf, wsq, CO, CI);
    demod_kernel<<<cdiv(8L * CO, 256), 256, 0, stream>>>(sbuf, wsq, dmbuf, CO, CI);
    int HW = H * W;
    if (HW >= 32) {
      dim3 g((unsigned)(HW / 32), (unsigned)(CO / 64), 8);
      conv_wmma_kernel<2><<<g, 128, 0, stream>>>(Abuf, xs, y, dmbuf, noise,
                                                 c.nw, c.bias, CO, CI, H, W);
    } else {
      dim3 g((unsigned)(HW / 16), (unsigned)(CO / 64), 8);
      conv_wmma_kernel<1><<<g, 128, 0, stream>>>(Abuf, xs, y, dmbuf, noise,
                                                 c.nw, c.bias, CO, CI, H, W);
    }
  };

  // ---- 2. conv1 @ 4x4 from const (batch-broadcast) ----
  run_style(conv1.modW, conv1.modB, sbuf, 512);
  scale_kernel<<<cdiv(8L * 512 * 16, 256), 256, 0, stream>>>(constv, 0, sbuf, xsbuf, 512, 16);
  run_conv(conv1, noises[0], xsbuf, actA, 512, 512, 4, 4);

  // ---- 3. to_rgb1 ----
  run_style(rgb1.modW, rgb1.modB, srgb, 512);
  rgbw_kernel<<<cdiv(8L * 3 * 512, 256), 256, 0, stream>>>(rgb1.w, srgb, wmrgb, 512);
  torgb_kernel<<<dim3(1, 8), 256, 0, stream>>>(actA, wmrgb, rgb1.bias,
                                               nullptr, skipA, 512, 4, 4);

  // ---- 4. progressive layers: res 8,16,32,64,128,256 ----
  const int chan[7] = {512, 512, 512, 512, 512, 256, 128}; // res 4..256
  float* skipPrev = skipA;
  float* skipFree = skipB;
  for (int j = 0; j < 6; ++j) {
    int res = 8 << j;
    int ci = chan[j], co = chan[j + 1];
    int H = res, W = res, HW = res * res;
    int Hh = res / 2;

    // convA: upsample(x) then 3x3 conv, ci -> co
    const SC& cA = convs[2 * j];
    run_style(cA.modW, cA.modB, sbuf, ci);
    up2scale_kernel<<<cdiv(8L * ci * HW, 256), 256, 0, stream>>>(actA, sbuf, xsbuf, ci, Hh, Hh);
    run_conv(cA, noises[2 * j + 1], xsbuf, actB, co, ci, H, W);

    // convB: 3x3 conv at same res, co -> co
    const SC& cB = convs[2 * j + 1];
    run_style(cB.modW, cB.modB, sbuf, co);
    scale_kernel<<<cdiv(8L * co * HW, 256), 256, 0, stream>>>(actB, co * HW, sbuf, xsbuf, co, HW);
    run_conv(cB, noises[2 * j + 2], xsbuf, actA, co, co, H, W);

    // to_rgb + upsampled skip
    const RG& rg = rgbs[j];
    run_style(rg.modW, rg.modB, srgb, co);
    rgbw_kernel<<<cdiv(8L * 3 * co, 256), 256, 0, stream>>>(rg.w, srgb, wmrgb, co);
    float* dst = (j == 5) ? (float*)d_out : skipFree;
    torgb_kernel<<<dim3(cdiv(HW, 256), 8), 256, 0, stream>>>(actA, wmrgb, rg.bias,
                                                             skipPrev, dst, co, H, W);
    skipFree = skipPrev;
    skipPrev = dst;
  }
}
